// ESA_23991687315605
// MI455X (gfx1250) — matrix operand order repaired and checked by reading, not compiled, not run on hardware
//
#include <hip/hip_runtime.h>
#include <hip/hip_bf16.h>

// Efficient attention for MI455X (gfx1250) — bf16 WMMA pipeline.
// All GEMMs via v_wmma_f32_16x16x32_bf16 (wave32, f32 accum).
// Activation tiles are staged into LDS pre-swizzled into B-fragment order so
// fragment loads are ds_load_b128 (no scalar u16 storms). Transpose-free tiles
// (weights, ctx^T) are staged with CDNA5 async global->LDS DMA (ASYNCcnt).

typedef __bf16 bf16;
typedef __attribute__((ext_vector_type(16))) __bf16 v16bf;
typedef __attribute__((ext_vector_type(8)))  __bf16 v8bf;
typedef __attribute__((ext_vector_type(8)))  float  v8f;

#define N_BATCH 8
#define C_DIM   512
#define HW_DIM  9216
#define HEADS   8
#define HEADC   64

#define QKV_MT  64      // block tile rows (output channels)
#define QKV_NT  128     // block tile cols (spatial)
#define KSTEP   32      // WMMA K per step
#define WSTR    48      // weight-tile LDS row stride, elems (96B = 6x16B)
#define LSTR    24      // per-lane fragment stride, elems (48B, 16B aligned)

// ---- CDNA5 async global->LDS (CDNA5 ISA) -------------
#define ASYNC_B128(ldsoff, gptr)                                          \
  asm volatile("global_load_async_to_lds_b128 %0, %1, off"                \
               :: "v"(ldsoff), "v"(gptr) : "memory")
#define WAIT_ASYNC() asm volatile("s_wait_asynccnt 0x0" ::: "memory")

static __device__ __forceinline__ unsigned lds_off(const void* p) {
  // Flat LDS aperture: addr[31:0] is the wave-relative LDS byte address.
  return (unsigned)(unsigned long long)p;
}

static __device__ __forceinline__ v8f wmma_bf16(v16bf a, v16bf b, v8f c) {
  return __builtin_amdgcn_wmma_f32_16x16x32_bf16(false, a, false, b, (short)0, c,
                                                 false, false);
}

static __device__ __forceinline__ v8f v8f_zero() {
  v8f z = {0.f, 0.f, 0.f, 0.f, 0.f, 0.f, 0.f, 0.f};
  return z;
}

// A fragment (16x32 bf16): lane L holds row M=(L&15);
// elem e -> K = khalf+e, khalf=16*(L>>4) (matches B contiguous order).
static __device__ __forceinline__ v16bf frag_a(const bf16* __restrict__ row, int lane) {
  const int khalf = (lane >> 4) * 16;
  union { v16bf v; bf16 h[16]; } u;
#pragma unroll
  for (int e = 0; e < 16; ++e) u.h[e] = row[khalf + e];
  return u.v;
}

// B fragment (32x16 bf16): lane L holds col N=(L&15), elem e -> K=16*(L>>4)+e.
// Caller hands a pointer to this lane's 16 consecutive K elements.
static __device__ __forceinline__ v16bf frag_b_contig(const bf16* __restrict__ p) {
  union { v16bf v; bf16 h[16]; } u;
#pragma unroll
  for (int e = 0; e < 16; ++e) u.h[e] = p[e];
  return u.v;
}

// ---------------------------------------------------------------- conversions
__global__ void k_f32_to_bf16(const float* __restrict__ in, bf16* __restrict__ out,
                              int count) {
  int i = blockIdx.x * blockDim.x + threadIdx.x;
  const int stride = gridDim.x * blockDim.x;
  for (; i < count; i += stride) out[i] = (bf16)in[i];
}

__global__ void k_zero_f32(float* __restrict__ p, int count) {
  int i = blockIdx.x * blockDim.x + threadIdx.x;
  const int stride = gridDim.x * blockDim.x;
  for (; i < count; i += stride) p[i] = 0.0f;
}

// ------------------------------------------------------- fused QKV projection
// Y_q = W_q * X + b_q, q in {K,Q,V}; per batch [512x512]*[512x9216].
// Grid: n(8)*mtiles(8)*ptiles(72), block 256 = 8 waves.
__global__ __launch_bounds__(256) void k_gemm_qkv(
    const bf16* __restrict__ xbf,
    const bf16* __restrict__ wk, const bf16* __restrict__ wq,
    const bf16* __restrict__ wv,
    const float* __restrict__ bk, const float* __restrict__ bq,
    const float* __restrict__ bv,
    bf16* __restrict__ ko, bf16* __restrict__ qo, bf16* __restrict__ vo)
{
  __shared__ __align__(16) bf16 Ws[3][QKV_MT][WSTR];       // A tiles (async DMA)
  __shared__ __align__(16) bf16 XsF[8 * 32 * LSTR];        // B tiles, frag order

  const int tid  = threadIdx.x;
  const int lane = tid & 31;
  const int wave = tid >> 5;

  const int b    = blockIdx.x;
  const int nIdx = b / (8 * 72);
  const int rem  = b % (8 * 72);
  const int mBlk = (rem / 72) * QKV_MT;
  const int pBlk = (rem % 72) * QKV_NT;

  const size_t tenOff = (size_t)nIdx * C_DIM * HW_DIM;
  const bf16* X = xbf + tenOff;
  const bf16* W[3]     = { wk, wq, wv };
  const float* bias[3] = { bk, bq, bv };
  bf16* out[3]         = { ko + tenOff, qo + tenOff, vo + tenOff };

  const int mt     = wave >> 1;        // 0..3 : 16-row tile inside 64
  const int ntBase = (wave & 1) * 4;   // 0 or 4 of 8 column tiles

  v8f acc[3][4];
#pragma unroll
  for (int q = 0; q < 3; ++q)
#pragma unroll
    for (int t = 0; t < 4; ++t) acc[q][t] = v8f_zero();

  for (int k0 = 0; k0 < C_DIM; k0 += KSTEP) {
    __syncthreads();
    // Async-DMA the three 64x32 weight tiles: one 16B chunk/thread/weight.
    {
      const int r = tid >> 2;            // row 0..63
      const int c = (tid & 3) * 8;       // k-offset 0/8/16/24
#pragma unroll
      for (int q = 0; q < 3; ++q) {
        const bf16* src = W[q] + (size_t)(mBlk + r) * C_DIM + k0 + c;
        ASYNC_B128(lds_off(&Ws[q][r][c]), src);
      }
    }
    // Stage X tile (32x128) swizzled into fragment order:
    // XsF[((p>>4)*32 + (r>>4)*16 + (p&15))*LSTR + (r&15)] = X[k0+r][pBlk+p]
#pragma unroll
    for (int j = 0; j < 2; ++j) {
      const int lin = tid * 16 + j * 8;
      const int r   = lin >> 7;          // k row 0..31
      const int cc  = lin & 127;         // p col (multiple of 8)
      const bf16* src = X + (size_t)(k0 + r) * HW_DIM + pBlk + cc;
      union { v8bf v; bf16 h[8]; } tmp;
      tmp.v = *(const v8bf*)src;
      const int e   = r & 15;
      const int lhi = (r >> 4) * 16;
#pragma unroll
      for (int jj = 0; jj < 8; ++jj) {
        const int p = cc + jj;
        XsF[((p >> 4) * 32 + lhi + (p & 15)) * LSTR + e] = tmp.h[jj];
      }
      if (k0 + KSTEP < C_DIM)
        __builtin_prefetch(src + (size_t)KSTEP * HW_DIM, 0, 3);
    }
    WAIT_ASYNC();
    __syncthreads();

    v16bf bfrag[4];
#pragma unroll
    for (int t = 0; t < 4; ++t)
      bfrag[t] = frag_b_contig(&XsF[((ntBase + t) * 32 + lane) * LSTR]);
#pragma unroll
    for (int q = 0; q < 3; ++q) {
      v16bf afrag = frag_a(&Ws[q][mt * 16 + (lane & 15)][0], lane);
#pragma unroll
      for (int t = 0; t < 4; ++t) acc[q][t] = wmma_bf16(afrag, bfrag[t], acc[q][t]);
    }
  }

  // Epilogue: bias + bf16 store. Lane owns col (L&15), rows v + 8*(L>>4).
#pragma unroll
  for (int q = 0; q < 3; ++q) {
    float b8[8];
#pragma unroll
    for (int v = 0; v < 8; ++v)
      b8[v] = bias[q][mBlk + mt * 16 + v + 8 * (lane >> 4)];
#pragma unroll
    for (int t = 0; t < 4; ++t) {
      union { v8f v; float f[8]; } a; a.v = acc[q][t];
      const int col = pBlk + (ntBase + t) * 16 + (lane & 15);
#pragma unroll
      for (int v = 0; v < 8; ++v) {
        const int row = mBlk + mt * 16 + v + 8 * (lane >> 4);
        out[q][(size_t)row * HW_DIM + col] = (bf16)(a.f[v] + b8[v]);
      }
    }
  }
}

// ------------------------------------------------- softmax over spatial (keys)
__global__ __launch_bounds__(256) void k_softmax_rows(bf16* __restrict__ data) {
  __shared__ float red[256];
  bf16* p = data + (size_t)blockIdx.x * HW_DIM;
  const int tid = threadIdx.x;
  float vals[36];
  float m = -3.0e38f;
#pragma unroll
  for (int i = 0; i < 36; ++i) {
    vals[i] = (float)p[tid + i * 256];
    m = fmaxf(m, vals[i]);
  }
  red[tid] = m; __syncthreads();
  for (int s = 128; s > 0; s >>= 1) {
    if (tid < s) red[tid] = fmaxf(red[tid], red[tid + s]);
    __syncthreads();
  }
  m = red[0]; __syncthreads();
  float sum = 0.f;
#pragma unroll
  for (int i = 0; i < 36; ++i) { vals[i] = __expf(vals[i] - m); sum += vals[i]; }
  red[tid] = sum; __syncthreads();
  for (int s = 128; s > 0; s >>= 1) {
    if (tid < s) red[tid] += red[tid + s];
    __syncthreads();
  }
  const float inv = 1.0f / red[0];
#pragma unroll
  for (int i = 0; i < 36; ++i) p[tid + i * 256] = (bf16)(vals[i] * inv);
}

// --------------------------------------------- softmax over head chans (query)
__global__ __launch_bounds__(256) void k_softmax_q(bf16* __restrict__ q) {
  const size_t idx = (size_t)blockIdx.x * blockDim.x + threadIdx.x;
  const size_t nh = idx / HW_DIM;
  const size_t p  = idx % HW_DIM;
  bf16* base = q + nh * (size_t)HEADC * HW_DIM + p;
  float v[HEADC];
  float m = -3.0e38f;
#pragma unroll
  for (int k = 0; k < HEADC; ++k) {
    v[k] = (float)base[(size_t)k * HW_DIM];
    m = fmaxf(m, v[k]);
  }
  float s = 0.f;
#pragma unroll
  for (int k = 0; k < HEADC; ++k) { v[k] = __expf(v[k] - m); s += v[k]; }
  const float inv = 1.0f / s;
#pragma unroll
  for (int k = 0; k < HEADC; ++k) base[(size_t)k * HW_DIM] = (bf16)(v[k] * inv);
}

// ---------------------------------------------------- context = keyS @ V^T
// Split-K over 9 chunks of 1024, global_atomic_add_f32 reduction.
// B loads are contiguous V rows (32B/lane) so no staging needed.
#define CTX_CHUNK 1024
__global__ __launch_bounds__(128) void k_context(
    const bf16* __restrict__ keyS, const bf16* __restrict__ vals,
    float* __restrict__ ctx)
{
  const int tid = threadIdx.x, lane = tid & 31, wave = tid >> 5;
  const int b     = blockIdx.x;
  const int nh    = b / (HW_DIM / CTX_CHUNK);
  const int chunk = b % (HW_DIM / CTX_CHUNK);
  const bf16* K = keyS + (size_t)nh * HEADC * HW_DIM;
  const bf16* V = vals + (size_t)nh * HEADC * HW_DIM;
  const int m0    = wave * 16;
  const int kbase = (lane >> 4) * 16;

  v8f acc[4];
#pragma unroll
  for (int t = 0; t < 4; ++t) acc[t] = v8f_zero();

  const int pEnd = (chunk + 1) * CTX_CHUNK;
  for (int p0 = chunk * CTX_CHUNK; p0 < pEnd; p0 += KSTEP) {
    v16bf a = frag_a(&K[(size_t)(m0 + (lane & 15)) * HW_DIM + p0], lane);
#pragma unroll
    for (int t = 0; t < 4; ++t) {
      v16bf bf = frag_b_contig(&V[(size_t)(t * 16 + (lane & 15)) * HW_DIM + p0 + kbase]);
      acc[t] = wmma_bf16(a, bf, acc[t]);
    }
  }

  float* cbase = ctx + (size_t)nh * HEADC * HEADC;
#pragma unroll
  for (int t = 0; t < 4; ++t) {
    union { v8f v; float f[8]; } a; a.v = acc[t];
    const int vcol = t * 16 + (lane & 15);
#pragma unroll
    for (int r = 0; r < 8; ++r) {
      const int krow = m0 + r + 8 * (lane >> 4);
      atomicAdd(&cbase[(size_t)krow * HEADC + vcol], a.f[r]);
    }
  }
}

// ctx f32 [nh][k][v] -> ctxT bf16 [nh][v][k]
__global__ void k_ctx_t(const float* __restrict__ ctx, bf16* __restrict__ ctxT) {
  const int idx = blockIdx.x * blockDim.x + threadIdx.x;
  const int nh = idx >> 12;
  const int k  = (idx >> 6) & 63;
  const int v  = idx & 63;
  ctxT[((size_t)nh << 12) + v * 64 + k] = (bf16)ctx[idx];
}

// --------------------------------------- attended = ctx^T @ queryS  (K = 64)
// Per (n,head,ptile): [64x64]*[64x128]. Block 256 = 8 waves, 4 tiles each.
// ctx^T tile via async DMA; Q tile staged once, swizzled to fragment order.
#define ASTR 72   // ctxT LDS row stride elems (144B = 9x16B)
__global__ __launch_bounds__(256) void k_attended(
    const bf16* __restrict__ ctxT, const bf16* __restrict__ queryS,
    bf16* __restrict__ att)
{
  __shared__ __align__(16) bf16 As[HEADC][ASTR];           // 64x64 ctx^T
  __shared__ __align__(16) bf16 QsF[2 * 8 * 32 * LSTR];    // [kstep][nt][lane][.]

  const int tid = threadIdx.x, lane = tid & 31, wave = tid >> 5;
  const int b    = blockIdx.x;        // nh*72 + ptile
  const int nh   = b / 72;
  const int pBlk = (b % 72) * QKV_NT;
  const bf16* A = ctxT + ((size_t)nh << 12);              // [v][k] row-major
  const bf16* Q = queryS + (size_t)nh * HEADC * HW_DIM;   // [k][p]
  const int n = nh >> 3, h = nh & 7;
  const int mt     = wave >> 1;
  const int ntBase = (wave & 1) * 4;

  // Async-DMA ctx^T tile: 64 rows x 32 elems = 256 16B chunks, 1/thread.
  {
    const int r = tid >> 2;
    const int c = (tid & 3) * 8;
    ASYNC_B128(lds_off(&As[r][c]), A + (size_t)r * HEADC + c);
  }
  // Stage Q tile (64x128) swizzled: 32 elems/thread = 4 x b128 reads.
#pragma unroll
  for (int j = 0; j < 4; ++j) {
    const int lin = tid * 32 + j * 8;
    const int r   = lin >> 7;          // k 0..63
    const int cc  = lin & 127;
    union { v8bf v; bf16 h[8]; } tmp;
    tmp.v = *(const v8bf*)(Q + (size_t)r * HW_DIM + pBlk + cc);
    const int ks  = r >> 5;
    const int e   = r & 15;
    const int lhi = ((r >> 4) & 1) * 16;
#pragma unroll
    for (int jj = 0; jj < 8; ++jj) {
      const int p = cc + jj;
      QsF[((ks * 8 + (p >> 4)) * 32 + lhi + (p & 15)) * LSTR + e] = tmp.h[jj];
    }
  }
  WAIT_ASYNC();
  __syncthreads();

  v8f acc[4];
#pragma unroll
  for (int t = 0; t < 4; ++t) acc[t] = v8f_zero();

#pragma unroll
  for (int s = 0; s < 2; ++s) {
    v16bf a = frag_a(&As[mt * 16 + (lane & 15)][s * KSTEP], lane);
#pragma unroll
    for (int t = 0; t < 4; ++t) {
      v16bf bf = frag_b_contig(&QsF[((s * 8 + ntBase + t) * 32 + lane) * LSTR]);
      acc[t] = wmma_bf16(a, bf, acc[t]);
    }
  }

  bf16* out = att + ((size_t)n * C_DIM + h * HEADC) * HW_DIM;
#pragma unroll
  for (int t = 0; t < 4; ++t) {
    union { v8f v; float f[8]; } a; a.v = acc[t];
    const int col = pBlk + (ntBase + t) * 16 + (lane & 15);
#pragma unroll
    for (int r = 0; r < 8; ++r) {
      const int row = mt * 16 + r + 8 * (lane >> 4);
      out[(size_t)row * HW_DIM + col] = (bf16)a.f[r];
    }
  }
}

// ------------------------------- output projection + bias + residual (f32 out)
__global__ __launch_bounds__(256) void k_gemm_out(
    const bf16* __restrict__ att, const bf16* __restrict__ wp,
    const float* __restrict__ bp, const float* __restrict__ xres,
    float* __restrict__ outp)
{
  __shared__ __align__(16) bf16 Ws[QKV_MT][WSTR];
  __shared__ __align__(16) bf16 XsF[8 * 32 * LSTR];

  const int tid  = threadIdx.x;
  const int lane = tid & 31;
  const int wave = tid >> 5;

  const int b    = blockIdx.x;
  const int nIdx = b / (8 * 72);
  const int rem  = b % (8 * 72);
  const int mBlk = (rem / 72) * QKV_MT;
  const int pBlk = (rem % 72) * QKV_NT;

  const bf16* X = att + (size_t)nIdx * C_DIM * HW_DIM;
  const int mt     = wave >> 1;
  const int ntBase = (wave & 1) * 4;

  v8f acc[4];
#pragma unroll
  for (int t = 0; t < 4; ++t) acc[t] = v8f_zero();

  for (int k0 = 0; k0 < C_DIM; k0 += KSTEP) {
    __syncthreads();
    {
      const int r = tid >> 2;
      const int c = (tid & 3) * 8;
      ASYNC_B128(lds_off(&Ws[r][c]), wp + (size_t)(mBlk + r) * C_DIM + k0 + c);
    }
#pragma unroll
    for (int j = 0; j < 2; ++j) {
      const int lin = tid * 16 + j * 8;
      const int r   = lin >> 7;
      const int cc  = lin & 127;
      const bf16* src = X + (size_t)(k0 + r) * HW_DIM + pBlk + cc;
      union { v8bf v; bf16 h[8]; } tmp;
      tmp.v = *(const v8bf*)src;
      const int e   = r & 15;
      const int lhi = (r >> 4) * 16;
#pragma unroll
      for (int jj = 0; jj < 8; ++jj) {
        const int p = cc + jj;
        XsF[((p >> 4) * 32 + lhi + (p & 15)) * LSTR + e] = tmp.h[jj];
      }
      if (k0 + KSTEP < C_DIM)
        __builtin_prefetch(src + (size_t)KSTEP * HW_DIM, 0, 3);
    }
    WAIT_ASYNC();
    __syncthreads();

    v16bf bfrag[4];
#pragma unroll
    for (int t = 0; t < 4; ++t)
      bfrag[t] = frag_b_contig(&XsF[((ntBase + t) * 32 + lane) * LSTR]);
    v16bf afrag = frag_a(&Ws[mt * 16 + (lane & 15)][0], lane);
#pragma unroll
    for (int t = 0; t < 4; ++t) acc[t] = wmma_bf16(afrag, bfrag[t], acc[t]);
  }

  float b8[8];
#pragma unroll
  for (int v = 0; v < 8; ++v)
    b8[v] = bp[mBlk + mt * 16 + v + 8 * (lane >> 4)];
#pragma unroll
  for (int t = 0; t < 4; ++t) {
    union { v8f v; float f[8]; } a; a.v = acc[t];
    const int col = pBlk + (ntBase + t) * 16 + (lane & 15);
#pragma unroll
    for (int v = 0; v < 8; ++v) {
      const int row = mBlk + mt * 16 + v + 8 * (lane >> 4);
      const size_t gi = ((size_t)nIdx * C_DIM + row) * HW_DIM + col;
      outp[gi] = a.f[v] + b8[v] + xres[gi];
    }
  }
}

// ---------------------------------------------------------------------- launch
extern "C" void kernel_launch(void* const* d_in, const int* in_sizes, int n_in,
                              void* d_out, int out_size, void* d_ws, size_t ws_size,
                              hipStream_t stream) {
  const float* x  = (const float*)d_in[0];
  const float* Wk = (const float*)d_in[1];
  const float* bk = (const float*)d_in[2];
  const float* Wq = (const float*)d_in[3];
  const float* bq = (const float*)d_in[4];
  const float* Wv = (const float*)d_in[5];
  const float* bv = (const float*)d_in[6];
  const float* Wp = (const float*)d_in[7];
  const float* bp = (const float*)d_in[8];
  float* out = (float*)d_out;

  const size_t tens = (size_t)N_BATCH * C_DIM * HW_DIM;
  const size_t wsz  = (size_t)C_DIM * C_DIM;
  const size_t ctxN = (size_t)N_BATCH * HEADS * HEADC * HEADC;

  char* ws = (char*)d_ws;
  size_t off = 0;
  auto take = [&](size_t bytes) -> char* {
    char* p = ws + off;
    off += (bytes + 255) & ~(size_t)255;
    return p;
  };
  bf16* xbf  = (bf16*)take(tens * 2);
  bf16* wkbf = (bf16*)take(wsz * 2);
  bf16* wqbf = (bf16*)take(wsz * 2);
  bf16* wvbf = (bf16*)take(wsz * 2);
  bf16* wpbf = (bf16*)take(wsz * 2);
  bf16* kbf  = (bf16*)take(tens * 2);
  bf16* qbf  = (bf16*)take(tens * 2);
  bf16* vbf  = (bf16*)take(tens * 2);
  bf16* attb = (bf16*)take(tens * 2);
  float* ctx = (float*)take(ctxN * 4);
  bf16* ctxT = (bf16*)take(ctxN * 2);

  k_f32_to_bf16<<<4096, 256, 0, stream>>>(x, xbf, (int)tens);
  k_f32_to_bf16<<<256, 256, 0, stream>>>(Wk, wkbf, (int)wsz);
  k_f32_to_bf16<<<256, 256, 0, stream>>>(Wq, wqbf, (int)wsz);
  k_f32_to_bf16<<<256, 256, 0, stream>>>(Wv, wvbf, (int)wsz);
  k_f32_to_bf16<<<256, 256, 0, stream>>>(Wp, wpbf, (int)wsz);

  k_gemm_qkv<<<N_BATCH * 8 * 72, 256, 0, stream>>>(
      xbf, wkbf, wqbf, wvbf, bk, bq, bv, kbf, qbf, vbf);

  k_softmax_rows<<<N_BATCH * C_DIM, 256, 0, stream>>>(kbf);
  k_softmax_q<<<(N_BATCH * HEADS * HW_DIM) / 256, 256, 0, stream>>>(qbf);

  k_zero_f32<<<1024, 256, 0, stream>>>(ctx, (int)ctxN);
  k_context<<<N_BATCH * HEADS * (HW_DIM / CTX_CHUNK), 128, 0, stream>>>(kbf, vbf, ctx);
  k_ctx_t<<<(int)(ctxN / 256), 256, 0, stream>>>(ctx, ctxT);

  k_attended<<<N_BATCH * HEADS * 72, 256, 0, stream>>>(ctxT, qbf, attb);

  k_gemm_out<<<N_BATCH * 8 * 72, 256, 0, stream>>>(attb, wpbf, bp, x, out);
}
